// RelativeAttention_56822417326644
// MI455X (gfx1250) — matrix operand order repaired and checked by reading, not compiled, not run on hardware
//
#include <hip/hip_runtime.h>
#include <hip/hip_bf16.h>

typedef __attribute__((ext_vector_type(16))) _Float16 v16h;
typedef __attribute__((ext_vector_type(8)))  _Float16 v8h;
typedef __attribute__((ext_vector_type(8)))  float    v8f;

#define AB    16      // batch
#define AN    512     // seq len
#define AD    1024    // model dim
#define AH    16      // heads
#define ADH   64      // dim head
#define ANF   3072    // 3*inner

// async-to-LDS builtin wants pointers to 16-byte int vectors with explicit AS
typedef int v4i __attribute__((vector_size(16)));
typedef __attribute__((address_space(1))) v4i v4i_as1;
typedef __attribute__((address_space(3))) v4i v4i_as3;

// ---- CDNA5 async copy: global -> LDS, 16 bytes per lane, tracked by ASYNCcnt ----
__device__ inline void async_cp16(const void* g, void* lds) {
#if __has_builtin(__builtin_amdgcn_global_load_async_to_lds_b128)
    __builtin_amdgcn_global_load_async_to_lds_b128(
        (v4i_as1*)(unsigned long long)g,
        (v4i_as3*)(unsigned)(unsigned long long)lds,  // low 32 bits = LDS byte offset
        0, 0);
#else
    asm volatile("global_load_async_to_lds_b128 %0, %1, off"
                 :: "v"((unsigned)(unsigned long long)lds), "v"(g)
                 : "memory");
#endif
}

template <int N>
__device__ inline void async_wait() {
#if __has_builtin(__builtin_amdgcn_s_wait_asynccnt)
    __builtin_amdgcn_s_wait_asynccnt(N);
#else
    asm volatile("s_wait_asynccnt %0" :: "i"(N) : "memory");
#endif
    asm volatile("" ::: "memory");   // compiler barrier: LDS now updated
}

__device__ inline v16h combine16(v8h lo, v8h hi) {
    v16h a;
#pragma unroll
    for (int e = 0; e < 8; ++e) { a[e] = lo[e]; a[e + 8] = hi[e]; }
    return a;
}

__device__ inline v8f wmma16(v16h a, v16h b, v8f c) {
    return __builtin_amdgcn_wmma_f32_16x16x32_f16(
        false, a, false, b, (short)0, c, false, false);
}

// wt[f*K + k] = (f16) w[k*Nf + f]   (transpose + convert to f16)
__global__ __launch_bounds__(256)
void wt_convert(const float* __restrict__ w, _Float16* __restrict__ wt,
                int K, int Nf) {
    long idx = (long)blockIdx.x * blockDim.x + threadIdx.x;
    long total = (long)K * Nf;
    if (idx >= total) return;
    int f = (int)(idx / K);
    int k = (int)(idx % K);
    wt[idx] = (_Float16)w[(size_t)k * Nf + f];
}

// Build an A fragment (16x32 f16) from an fp32 row: halves kb+b0+[0..7], kb+16+b0+[0..7]
__device__ inline v16h a_frag_f32(const float* arow, int kb, int b0) {
    const float4 f0 = *(const float4*)(arow + kb + b0);
    const float4 f1 = *(const float4*)(arow + kb + b0 + 4);
    const float4 f2 = *(const float4*)(arow + kb + 16 + b0);
    const float4 f3 = *(const float4*)(arow + kb + 16 + b0 + 4);
    v16h a;
    a[0]=(_Float16)f0.x; a[1]=(_Float16)f0.y; a[2]=(_Float16)f0.z; a[3]=(_Float16)f0.w;
    a[4]=(_Float16)f1.x; a[5]=(_Float16)f1.y; a[6]=(_Float16)f1.z; a[7]=(_Float16)f1.w;
    a[8]=(_Float16)f2.x; a[9]=(_Float16)f2.y; a[10]=(_Float16)f2.z; a[11]=(_Float16)f2.w;
    a[12]=(_Float16)f3.x; a[13]=(_Float16)f3.y; a[14]=(_Float16)f3.z; a[15]=(_Float16)f3.w;
    return a;
}

// qkv = x @ w_qkv + b_qkv ; scatter into Q(bh,n,d)*scale, K(bh,n,d), Vt(bh,d,n), f16.
// One wave computes a 32x64 tile: 2 A-frags share each B-frag -> 8 WMMA per k-chunk.
__global__ __launch_bounds__(32)
void qkv_gemm(const float* __restrict__ x, const _Float16* __restrict__ wt,
              const float* __restrict__ bias,
              _Float16* __restrict__ Qo, _Float16* __restrict__ Ko,
              _Float16* __restrict__ Vt) {
    const int lane = threadIdx.x;
    const int r  = lane & 15;
    const int hi = lane >> 4;
    const int b0 = hi * 8;
    const int mbase = blockIdx.x * 32;
    const int nbase = blockIdx.y * 64;

    v8f zero = {};
    v8f acc[2][4];
#pragma unroll
    for (int rt = 0; rt < 2; ++rt)
#pragma unroll
        for (int t = 0; t < 4; ++t) acc[rt][t] = zero;

    const float* arow0 = x + (size_t)(mbase + r) * AD;
    const float* arow1 = arow0 + (size_t)16 * AD;
    for (int kb = 0; kb < AD; kb += 32) {
        v16h a0 = a_frag_f32(arow0, kb, b0);
        v16h a1 = a_frag_f32(arow1, kb, b0);
#pragma unroll
        for (int tt = 0; tt < 4; ++tt) {
            const _Float16* bp =
                wt + (size_t)(nbase + tt * 16 + r) * AD + kb + hi * 8;
            v8h blo = *(const v8h*)bp;
            v8h bhi = *(const v8h*)(bp + 16);
            v16h bv = combine16(blo, bhi);
            acc[0][tt] = wmma16(a0, bv, acc[0][tt]);
            acc[1][tt] = wmma16(a1, bv, acc[1][tt]);
        }
    }
#pragma unroll
    for (int tt = 0; tt < 4; ++tt) {
        const int f    = nbase + tt * 16 + r;     // output feature (C col = lane&15)
        const int sel  = f >> 10;                 // 0:q 1:k 2:v  (uniform per tile)
        const int fr   = f & 1023;
        const int head = fr >> 6;                 // uniform per tile
        const int d    = fr & 63;
        const float bb = bias[f];
        if (sel == 0) {
#pragma unroll
            for (int rt = 0; rt < 2; ++rt)
#pragma unroll
                for (int g = 0; g < 8; ++g) {
                    const int m = mbase + rt * 16 + g + hi * 8;
                    const int bh = (m >> 9) * AH + head, n = m & 511;
                    Qo[((size_t)bh * AN + n) * ADH + d] =
                        (_Float16)((acc[rt][tt][g] + bb) * 0.125f);
                }
        } else if (sel == 1) {
#pragma unroll
            for (int rt = 0; rt < 2; ++rt)
#pragma unroll
                for (int g = 0; g < 8; ++g) {
                    const int m = mbase + rt * 16 + g + hi * 8;
                    const int bh = (m >> 9) * AH + head, n = m & 511;
                    Ko[((size_t)bh * AN + n) * ADH + d] =
                        (_Float16)(acc[rt][tt][g] + bb);
                }
        } else {
#pragma unroll
            for (int rt = 0; rt < 2; ++rt)
#pragma unroll
                for (int g = 0; g < 8; ++g) {
                    const int m = mbase + rt * 16 + g + hi * 8;
                    const int bh = (m >> 9) * AH + head, n = m & 511;
                    Vt[((size_t)bh * ADH + d) * AN + n] =
                        (_Float16)(acc[rt][tt][g] + bb);
                }
        }
    }
}

// one wave per (bh, 16-row query tile): S = QK^T(+bias) -> softmax -> P@V -> AO f16.
// K tiles are double-buffered into LDS with CDNA5 async-to-LDS copies (ASYNCcnt).
__global__ __launch_bounds__(32)
void attention(const _Float16* __restrict__ Q, const _Float16* __restrict__ K,
               const _Float16* __restrict__ Vt, const float* __restrict__ rel,
               _Float16* __restrict__ AO) {
    const int lane = threadIdx.x;
    const int r  = lane & 15;
    const int hi = lane >> 4;
    const int b0 = hi * 8;
    const int qbase = blockIdx.x * 16;
    const int bh    = blockIdx.y;
    const int head  = bh & 15;
    const int b     = bh >> 4;

    __shared__ float    sc[16][AN];          // 32 KB score block
    __shared__ _Float16 pr[16][AN];          // 16 KB probabilities (A-frag layout friendly)
    __shared__ _Float16 Ks[2][16][ADH];      //  4 KB double-buffered K tiles
    __shared__ float    rowsum[16];

    const _Float16* kbase = K + (size_t)bh * AN * ADH;

    // Q fragments (K-dim = 64 -> two chunks of 32); Q pre-scaled by dh^-0.5
    const _Float16* qrow = Q + ((size_t)bh * AN + qbase + r) * ADH;
    v16h aq[2];
#pragma unroll
    for (int c2 = 0; c2 < 2; ++c2) {
        v8h lo = *(const v8h*)(qrow + c2 * 32 + b0);
        v8h hv = *(const v8h*)(qrow + c2 * 32 + 16 + b0);
        aq[c2] = combine16(lo, hv);
    }

    // stage K tile 0 (2 KB contiguous) into LDS: 4 x b128 async per lane
#pragma unroll
    for (int c = 0; c < 4; ++c) {
        const int idx = c * 32 + lane;
        async_cp16(kbase + (size_t)idx * 8, (char*)&Ks[0][0][0] + idx * 16);
    }

    v8f zero = {};
    for (int j = 0; j < 32; ++j) {
        const int buf = j & 1;
        if (j + 1 < 32) {
            const _Float16* knext = kbase + (size_t)(j + 1) * 16 * ADH;
#pragma unroll
            for (int c = 0; c < 4; ++c) {
                const int idx = c * 32 + lane;
                async_cp16(knext + (size_t)idx * 8,
                           (char*)&Ks[buf ^ 1][0][0] + idx * 16);
            }
            async_wait<4>();    // tile j complete; tile j+1 still in flight
        } else {
            async_wait<0>();
        }
        v8f s = zero;
#pragma unroll
        for (int c2 = 0; c2 < 2; ++c2) {
            // B = K^T: lane holds key column jbase+r, kdims c2*32 + hi*8 + [0..7], c2*32 + 16 + hi*8 + [0..7]
            v8h lo = *(const v8h*)&Ks[buf][r][c2 * 32 + hi * 8];
            v8h hv = *(const v8h*)&Ks[buf][r][c2 * 32 + 16 + hi * 8];
            s = wmma16(aq[c2], combine16(lo, hv), s);
        }
        const int jj = j * 16 + r;
#pragma unroll
        for (int g = 0; g < 8; ++g) {
            const int row = g + hi * 8;
            const int i   = qbase + row;
            const float bias = rel[(size_t)(i - jj + (AN - 1)) * AH + head];
            sc[row][jj] = s[g] + bias;
        }
    }
    __syncthreads();

    // softmax over 512: lanes (r, r+16) split row r into halves, combine via shfl_xor
    {
        const int cb = hi * 256;
        float m = -1e30f;
        for (int t = 0; t < 256; ++t) m = fmaxf(m, sc[r][cb + t]);
        m = fmaxf(m, __shfl_xor(m, 16, 32));
        float sum = 0.f;
        for (int t = 0; t < 256; ++t) {
            float e = __expf(sc[r][cb + t] - m);
            pr[r][cb + t] = (_Float16)e;
            sum += e;
        }
        sum += __shfl_xor(sum, 16, 32);
        if (hi == 0) rowsum[r] = sum;
    }
    __syncthreads();

    // O = P @ V : 4 accumulators (dh tiles), 16 K-chunks of 32 keys
    v8f o[4];
#pragma unroll
    for (int t = 0; t < 4; ++t) o[t] = zero;
    for (int kc = 0; kc < 16; ++kc) {
        const int kb = kc * 32;
        v8h plo = *(const v8h*)&pr[r][kb + b0];
        v8h phi = *(const v8h*)&pr[r][kb + 16 + b0];
        v16h ap = combine16(plo, phi);
#pragma unroll
        for (int tt = 0; tt < 4; ++tt) {
            // B = V: lane holds dh column tt*16+r; keys contiguous in Vt
            const _Float16* vp =
                Vt + ((size_t)bh * ADH + tt * 16 + r) * AN + kb + hi * 8;
            if (kc + 1 < 16) __builtin_prefetch(vp + 32, 0, 1);
            v8h lo = *(const v8h*)vp;
            v8h hv = *(const v8h*)(vp + 16);
            o[tt] = wmma16(ap, combine16(lo, hv), o[tt]);
        }
    }
#pragma unroll
    for (int tt = 0; tt < 4; ++tt) {
#pragma unroll
        for (int g = 0; g < 8; ++g) {
            const int row = g + hi * 8;
            const float val = o[tt][g] / rowsum[row];
            const int token = qbase + row;
            AO[((size_t)(b * AN + token)) * AD + head * ADH + tt * 16 + r] =
                (_Float16)val;
        }
    }
}

// out = AO @ w_out + b_out (fp32 output); 32x64 tile per wave
__global__ __launch_bounds__(32)
void out_proj(const _Float16* __restrict__ AO, const _Float16* __restrict__ wt,
              const float* __restrict__ bias, float* __restrict__ out) {
    const int lane = threadIdx.x;
    const int r  = lane & 15;
    const int hi = lane >> 4;
    const int b0 = hi * 8;
    const int mbase = blockIdx.x * 32;
    const int nbase = blockIdx.y * 64;

    v8f zero = {};
    v8f acc[2][4];
#pragma unroll
    for (int rt = 0; rt < 2; ++rt)
#pragma unroll
        for (int t = 0; t < 4; ++t) acc[rt][t] = zero;

    const _Float16* arow0 = AO + (size_t)(mbase + r) * AD;
    const _Float16* arow1 = arow0 + (size_t)16 * AD;
    for (int kb = 0; kb < AD; kb += 32) {
        v16h a0 = combine16(*(const v8h*)(arow0 + kb + b0),
                            *(const v8h*)(arow0 + kb + 16 + b0));
        v16h a1 = combine16(*(const v8h*)(arow1 + kb + b0),
                            *(const v8h*)(arow1 + kb + 16 + b0));
#pragma unroll
        for (int tt = 0; tt < 4; ++tt) {
            const _Float16* bp =
                wt + (size_t)(nbase + tt * 16 + r) * AD + kb + hi * 8;
            v8h blo = *(const v8h*)bp;
            v8h bhi = *(const v8h*)(bp + 16);
            v16h bv = combine16(blo, bhi);
            acc[0][tt] = wmma16(a0, bv, acc[0][tt]);
            acc[1][tt] = wmma16(a1, bv, acc[1][tt]);
        }
    }
#pragma unroll
    for (int tt = 0; tt < 4; ++tt) {
        const int f = nbase + tt * 16 + r;
        const float bb = bias[f];
#pragma unroll
        for (int rt = 0; rt < 2; ++rt)
#pragma unroll
            for (int g = 0; g < 8; ++g) {
                const int m = mbase + rt * 16 + g + hi * 8;
                out[(size_t)m * AD + f] = acc[rt][tt][g] + bb;
            }
    }
}

extern "C" void kernel_launch(void* const* d_in, const int* in_sizes, int n_in,
                              void* d_out, int out_size, void* d_ws, size_t ws_size,
                              hipStream_t stream) {
    const float* x     = (const float*)d_in[0];
    const float* w_qkv = (const float*)d_in[1];
    const float* b_qkv = (const float*)d_in[2];
    const float* rel   = (const float*)d_in[3];
    const float* w_out = (const float*)d_in[4];
    const float* b_out = (const float*)d_in[5];
    float* out = (float*)d_out;

    char* ws = (char*)d_ws;
    const size_t sz_wtq = (size_t)ANF * AD * sizeof(_Float16);   // 6.0 MB
    const size_t sz_wto = (size_t)AD * AD * sizeof(_Float16);    // 2.0 MB
    const size_t sz_hds = (size_t)AB * AH * AN * ADH * sizeof(_Float16); // 16 MB each

    _Float16* wtq = (_Float16*)ws;                 ws += sz_wtq;
    _Float16* wto = (_Float16*)ws;                 ws += sz_wto;
    _Float16* Qb  = (_Float16*)ws;                 ws += sz_hds;
    _Float16* Kb  = (_Float16*)ws;                 ws += sz_hds;
    _Float16* Vt  = (_Float16*)ws;                 ws += sz_hds;
    _Float16* AO  = (_Float16*)ws;

    {
        long n = (long)AD * ANF;
        wt_convert<<<(unsigned)((n + 255) / 256), 256, 0, stream>>>(w_qkv, wtq, AD, ANF);
    }
    {
        long n = (long)AD * AD;
        wt_convert<<<(unsigned)((n + 255) / 256), 256, 0, stream>>>(w_out, wto, AD, AD);
    }
    // M = B*N = 8192 rows -> 256 row-tiles of 32; Nf = 3072 -> 48 col-tiles of 64
    qkv_gemm<<<dim3(256, 48), 32, 0, stream>>>(x, wtq, b_qkv, Qb, Kb, Vt);
    // 32 query tiles x 256 (b,h) pairs
    attention<<<dim3(AN / 16, AB * AH), 32, 0, stream>>>(Qb, Kb, Vt, rel, AO);
    // 256 row-tiles of 32 x 16 col-tiles of 64
    out_proj<<<dim3(256, AD / 64), 32, 0, stream>>>(AO, wto, b_out, out);
}
